// FBFETGRU_50955492000022
// MI455X (gfx1250) — matrix operand order repaired and checked by reading, not compiled, not run on hardware
//
#include <hip/hip_runtime.h>
#include <stdint.h>

// Problem constants (match reference)
#define BB 64
#define TT 2048
#define II 16
#define HH 256
#define G3 768          // 3*H
#define BT (BB*TT)      // 131072 rows

typedef __bf16        bf16x16 __attribute__((ext_vector_type(16)));
typedef float         f32x8   __attribute__((ext_vector_type(8)));
typedef unsigned int  u32x4   __attribute__((ext_vector_type(4)));

union FragAB { bf16x16 v; u32x4 u[2]; };

__device__ __forceinline__ unsigned short f2bf(float f) {
    unsigned u = __builtin_bit_cast(unsigned, f);
    u += 0x7fffu + ((u >> 16) & 1u);          // round-to-nearest-even
    return (unsigned short)(u >> 16);
}
__device__ __forceinline__ float bf2f(unsigned short h) {
    unsigned u = ((unsigned)h) << 16;
    return __builtin_bit_cast(float, u);
}
__device__ __forceinline__ float sigmoid_(float x) { return 1.f / (1.f + __expf(-x)); }
__device__ __forceinline__ float tanh_(float x) {
    x = fminf(15.f, fmaxf(-15.f, x));
    float e = __expf(2.f * x);
    return (e - 1.f) / (e + 1.f);
}
__device__ __forceinline__ void async_cp16(unsigned lds_off, const void* g) {
    asm volatile("global_load_async_to_lds_b128 %0, %1, off"
                 :: "v"(lds_off), "v"(g) : "memory");
}
__device__ __forceinline__ void wait_async0() {
    asm volatile("s_wait_asynccnt 0x0" ::: "memory");
}

// -------------------------------------------------------------------------
// Pack W [N][K] f32 row-major -> bf16 WMMA B-fragment tiles.
// Tile (kt, nt) covers B[k=kt*32..+31][n=nt*16..+15], B[k][n] = W[n][k].
// Within a tile (512 bf16, 1KB): lane l owns column n = nt*16 + (l&15),
// holds 16 consecutive k starting at kt*32 + (l>>4)*16 (contiguous k order, same as the A loads).
// -------------------------------------------------------------------------
__global__ void k_pack_b(const float* __restrict__ W, unsigned short* __restrict__ out,
                         int N, int K) {
    int o = blockIdx.x * 256 + threadIdx.x;
    if (o >= N * K) return;
    int tile = o >> 9, within = o & 511;
    int lane = within >> 4, v = within & 15;
    int ntiles = N >> 4;
    int nt = tile % ntiles, kt = tile / ntiles;
    int n = nt * 16 + (lane & 15);
    int k = kt * 32 + (lane >> 4) * 16 + v;
    out[o] = f2bf(W[n * K + k]);
}

// -------------------------------------------------------------------------
// Layer-0 input gates: xg[bt][g] = bih0[g] + sum_i x[bt][i]*Wih0[g][i]  (K=16)
// -------------------------------------------------------------------------
__global__ void k_xg0(const float* __restrict__ x, const float* __restrict__ Wih,
                      const float* __restrict__ bih, float* __restrict__ xg) {
    long long o = (long long)blockIdx.x * 256 + threadIdx.x;
    int g = (int)(o % G3);
    long long bt = o / G3;
    const float* xr = x + bt * II;
    const float* wr = Wih + g * II;
    float acc = bih[g];
#pragma unroll
    for (int i = 0; i < II; ++i) acc += xr[i] * wr[i];
    xg[o] = acc;
}

// -------------------------------------------------------------------------
// Persistent GRU scan, batch-parallel: grid=4 WGs x 16 batch rows each
// (sequences are independent across batch -> no cross-WG communication).
// Block = 512 (16 waves). Wave j owns H-column tile j for ALL 3 gates of
// its WG's single 16-row M tile -> gate fusion entirely in registers.
// Whh (packed bf16): kt-slices 0..4 staged once to LDS by async copies,
// kt-slices 5..7 held in VGPRs across the whole scan. h kept in LDS (8KB)
// for the WMMA A operand; h_prev kept in registers for the update.
// The kt loop is software-pipelined (prefetch kt+1 fragments before the
// kt WMMAs) so waits become s_wait_dscnt N>0 instead of full serialization;
// the kt==7 prefetch wraps to kt=0's (t-invariant) B so only the A fragment
// is on the critical path after the step barrier.
// LDS: 240KB weights + 8KB h = 248KB (< 320KB/WGP).
// -------------------------------------------------------------------------
#define KT_LDS 5
#define B_LDS_BYTES (KT_LDS * 48 * 1024)              // 245760
#define SCAN_LDS (B_LDS_BYTES + 16 * 256 * 2)         // + h: 253952

__global__ void __launch_bounds__(512)
__attribute__((amdgpu_waves_per_eu(4)))
k_scan(const float* __restrict__ xg, const unsigned short* __restrict__ Bp,
       const float* __restrict__ bhh, unsigned short* __restrict__ hout) {
    extern __shared__ char smem[];
    unsigned short* hS = (unsigned short*)(smem + B_LDS_BYTES);   // [16][256] bf16

    const int tid   = threadIdx.x;
    const int lane  = tid & 31;
    const int j     = tid >> 5;          // wave id = H-column tile (0..15)
    const int lh    = lane >> 4;         // lane half
    const int l15   = lane & 15;
    const int wgrow = blockIdx.x * 16;   // first batch row of this WG
    const int hh    = j * 16 + l15;      // this lane's H column

    // ---- stage Whh kt=0..4 (240KB) into LDS with async copies ----
    {
        const char* src = (const char*)Bp;
        for (int c = tid; c < B_LDS_BYTES / 16; c += 512)
            async_cp16((unsigned)(c * 16), src + (size_t)c * 16);
    }
    // ---- kt=5..7 B fragments live in registers for the whole scan ----
    FragAB Breg[3][3];   // [gate][kt-5]
#pragma unroll
    for (int g = 0; g < 3; ++g)
#pragma unroll
        for (int kk = 0; kk < 3; ++kk) {
            const u32x4* p = (const u32x4*)((const char*)Bp +
                (size_t)(((KT_LDS + kk) * 48) + (g * 16 + j)) * 1024 + lane * 32);
            Breg[g][kk].u[0] = p[0];
            Breg[g][kk].u[1] = p[1];
        }

    // per-lane recurrent biases (constant over t)
    const float bR = bhh[hh], bZ = bhh[256 + hh], bN = bhh[512 + hh];

    for (int i = tid; i < 16 * 256; i += 512) hS[i] = 0;   // h0 = 0
    float hprev[8];
#pragma unroll
    for (int v = 0; v < 8; ++v) hprev[v] = 0.f;

    wait_async0();
    __syncthreads();

    // fragment fetch helpers
    auto loadA = [&](int kt, FragAB& dst) {
        int kb = kt * 32 + lh * 16;
        dst.u[0] = *(const u32x4*)((const char*)hS + (l15 * 256 + kb) * 2);
        dst.u[1] = *(const u32x4*)((const char*)hS + (l15 * 256 + kb + 8) * 2);
    };
    auto loadB = [&](int kt, int g, FragAB& dst) {
        if (kt < KT_LDS) {
            const u32x4* p = (const u32x4*)(smem +
                (size_t)(kt * 48 + g * 16 + j) * 1024 + lane * 32);
            dst.u[0] = p[0];
            dst.u[1] = p[1];
        } else {
            dst = Breg[g][kt - KT_LDS];
        }
    };

    // prime xg registers for t = 0 (per lane: 3 gates x 8 rows)
    float xc[3][8];
#pragma unroll
    for (int v = 0; v < 8; ++v) {
        size_t row = (size_t)(wgrow + lh * 8 + v) * TT;   // * T, t=0
#pragma unroll
        for (int g = 0; g < 3; ++g) xc[g][v] = xg[row * G3 + g * 256 + hh];
    }

    // B double-buffer primed with kt=0 (t-invariant -> persists across steps)
    FragAB Bcur[3];
#pragma unroll
    for (int g = 0; g < 3; ++g) loadB(0, g, Bcur[g]);

    for (int t = 0; t < TT; ++t) {
        // ---- gh = h @ Whh^T, one 16-row M tile, 3 gate tiles per wave ----
        f32x8 acc[3];
#pragma unroll
        for (int g = 0; g < 3; ++g)
#pragma unroll
            for (int v = 0; v < 8; ++v) acc[g][v] = 0.f;

        FragAB Acur;
        loadA(0, Acur);

#pragma unroll
        for (int kt = 0; kt < 8; ++kt) {
            const int ktn = (kt + 1) & 7;            // wraps to 0 at kt==7
            FragAB Anext, Bnext[3];
            if (kt < 7) loadA(ktn, Anext);           // prefetch next A
#pragma unroll
            for (int g = 0; g < 3; ++g) loadB(ktn, g, Bnext[g]);   // prefetch next B
#pragma unroll
            for (int g = 0; g < 3; ++g)
                acc[g] = __builtin_amdgcn_wmma_f32_16x16x32_bf16(
                    false, Acur.v, false, Bcur[g].v, (short)0, acc[g], false, false);
            if (kt < 7) Acur = Anext;
#pragma unroll
            for (int g = 0; g < 3; ++g) Bcur[g] = Bnext[g];
        }
        __syncthreads();   // all A reads of h(t-1) done before h(t) writes

        // ---- issue xg prefetch for t+1 early (hide HBM latency) ----
        float xn_[3][8];
#pragma unroll
        for (int v = 0; v < 8; ++v)
#pragma unroll
            for (int g = 0; g < 3; ++g) xn_[g][v] = 0.f;
        if (t + 1 < TT) {
#pragma unroll
            for (int v = 0; v < 8; ++v) {
                size_t row = ((size_t)(wgrow + lh * 8 + v) * TT + (t + 1));
#pragma unroll
                for (int g = 0; g < 3; ++g) xn_[g][v] = xg[row * G3 + g * 256 + hh];
            }
        }

        // ---- gate fusion entirely in registers ----
#pragma unroll
        for (int v = 0; v < 8; ++v) {
            int m = lh * 8 + v;                      // local row, matches C/D layout
            float r = sigmoid_(xc[0][v] + acc[0][v] + bR);
            float z = sigmoid_(xc[1][v] + acc[1][v] + bZ);
            float n = tanh_(xc[2][v] + r * (acc[2][v] + bN));
            float hnew = (1.f - z) * n + z * hprev[v];
            hprev[v] = hnew;
            unsigned short hb = f2bf(hnew);
            hS[m * 256 + hh] = hb;
            hout[((size_t)(wgrow + m) * TT + t) * HH + hh] = hb;
        }
#pragma unroll
        for (int v = 0; v < 8; ++v)
#pragma unroll
            for (int g = 0; g < 3; ++g) xc[g][v] = xn_[g][v];

        __syncthreads();   // h(t) visible before next step's A reads
    }
}

// -------------------------------------------------------------------------
// Bulk GEMM: out[M][N] = act(A[M][K](bf16) @ B^T + bias).  B pre-packed.
// Workgroup: 256 threads / 8 waves, covers 256 rows x 32 cols.
// A block (256 x K bf16, contiguous) staged to LDS with async copies.
// -------------------------------------------------------------------------
template <bool RELU, bool OUTBF16>
__global__ void __launch_bounds__(256) k_gemm(const unsigned short* __restrict__ A,
                                              const unsigned short* __restrict__ Bp,
                                              const float* __restrict__ bias,
                                              void* __restrict__ outv,
                                              int M, int N, int K) {
    extern __shared__ char smem[];   // 256 * K bf16
    const int tid = threadIdx.x, lane = tid & 31, wave = tid >> 5;
    const int wn = wave & 1, wg = wave >> 1;   // wg: 4 m-tiles each
    const int mblock = blockIdx.x;             // M/256
    const int nblock = blockIdx.y;             // N/32

    {
        const char* gsrc = (const char*)(A + (size_t)mblock * 256 * K);
        const int chunks = (256 * K * 2) / 16;
        for (int c = tid; c < chunks; c += 256)
            async_cp16((unsigned)(c * 16), gsrc + (size_t)c * 16);
        wait_async0();
    }
    __syncthreads();

    f32x8 acc[4];
#pragma unroll
    for (int a = 0; a < 4; ++a)
#pragma unroll
        for (int v = 0; v < 8; ++v) acc[a][v] = 0.f;

    const int ktmax = K >> 5;
    for (int kt = 0; kt < ktmax; ++kt) {
        FragAB Bf;
        const u32x4* pb = (const u32x4*)((const char*)Bp +
                          (size_t)(kt * (N >> 4) + nblock * 2 + wn) * 1024 + lane * 32);
        Bf.u[0] = pb[0];
        Bf.u[1] = pb[1];
#pragma unroll
        for (int a = 0; a < 4; ++a) {
            int m  = (wg * 4 + a) * 16 + (lane & 15);
            int kb = kt * 32 + (lane >> 4) * 16;
            FragAB Af;
            Af.u[0] = *(const u32x4*)(smem + (m * K + kb) * 2);
            Af.u[1] = *(const u32x4*)(smem + (m * K + kb + 8) * 2);
            acc[a] = __builtin_amdgcn_wmma_f32_16x16x32_bf16(
                false, Af.v, false, Bf.v, (short)0, acc[a], false, false);
        }
    }

    int ncol = nblock * 32 + wn * 16 + (lane & 15);
    float bv = bias[ncol];
#pragma unroll
    for (int a = 0; a < 4; ++a) {
        int mrow0 = mblock * 256 + (wg * 4 + a) * 16 + (lane >> 4) * 8;
#pragma unroll
        for (int v = 0; v < 8; ++v) {
            float val = acc[a][v] + bv;
            if (RELU) val = fmaxf(val, 0.f);
            size_t o = (size_t)(mrow0 + v) * N + ncol;
            if (OUTBF16) ((unsigned short*)outv)[o] = f2bf(val);
            else         ((float*)outv)[o] = val;
        }
    }
}

// -------------------------------------------------------------------------
// Head: out[row] = relu(b2 + sum_h r[row][h] * W2[h]); wave per row.
// -------------------------------------------------------------------------
__global__ void __launch_bounds__(256) k_head(const unsigned short* __restrict__ r,
                                              const float* __restrict__ W2,
                                              const float* __restrict__ b2,
                                              float* __restrict__ out, int M) {
    int row  = blockIdx.x * 8 + (threadIdx.x >> 5);
    int lane = threadIdx.x & 31;
    if (row >= M) return;
    u32x4 rv = *(const u32x4*)(r + (size_t)row * 256 + lane * 8);
    const unsigned short* rh = (const unsigned short*)&rv;
    float s = 0.f;
#pragma unroll
    for (int j = 0; j < 8; ++j) s += bf2f(rh[j]) * W2[lane * 8 + j];
#pragma unroll
    for (int off = 16; off > 0; off >>= 1) s += __shfl_down(s, off, 32);
    if (lane == 0) out[row] = fmaxf(s + b2[0], 0.f);
}

// -------------------------------------------------------------------------
extern "C" void kernel_launch(void* const* d_in, const int* in_sizes, int n_in,
                              void* d_out, int out_size, void* d_ws, size_t ws_size,
                              hipStream_t stream) {
    const float* x    = (const float*)d_in[0];
    const float* Wih0 = (const float*)d_in[1];
    const float* Whh0 = (const float*)d_in[2];
    const float* bih0 = (const float*)d_in[3];
    const float* bhh0 = (const float*)d_in[4];
    const float* Wih1 = (const float*)d_in[5];
    const float* Whh1 = (const float*)d_in[6];
    const float* bih1 = (const float*)d_in[7];
    const float* bhh1 = (const float*)d_in[8];
    const float* W1   = (const float*)d_in[9];
    const float* b1   = (const float*)d_in[10];
    const float* W2   = (const float*)d_in[11];
    const float* b2   = (const float*)d_in[12];
    float* out = (float*)d_out;

    char* ws = (char*)d_ws;
    size_t off = 0;
    float*          xg    = (float*)(ws + off);          off += (size_t)BT * G3 * 4;   // 402MB
    unsigned short* h1    = (unsigned short*)(ws + off); off += (size_t)BT * HH * 2;   // 67MB
    unsigned short* h2    = (unsigned short*)(ws + off); off += (size_t)BT * HH * 2;
    unsigned short* rbuf  = (unsigned short*)(ws + off); off += (size_t)BT * HH * 2;
    unsigned short* whh0p = (unsigned short*)(ws + off); off += (size_t)G3 * HH * 2;
    unsigned short* whh1p = (unsigned short*)(ws + off); off += (size_t)G3 * HH * 2;
    unsigned short* wih1p = (unsigned short*)(ws + off); off += (size_t)G3 * HH * 2;
    unsigned short* w1p   = (unsigned short*)(ws + off); off += (size_t)HH * HH * 2;

    auto* gemm_f32  = k_gemm<false, false>;
    auto* gemm_relu = k_gemm<true, true>;
    hipFuncSetAttribute(reinterpret_cast<const void*>(k_scan),
                        hipFuncAttributeMaxDynamicSharedMemorySize, SCAN_LDS);
    hipFuncSetAttribute(reinterpret_cast<const void*>(gemm_f32),
                        hipFuncAttributeMaxDynamicSharedMemorySize, 256 * 256 * 2);
    hipFuncSetAttribute(reinterpret_cast<const void*>(gemm_relu),
                        hipFuncAttributeMaxDynamicSharedMemorySize, 256 * 256 * 2);

    // pack recurrent / dense weights into WMMA B-tile bf16 layout
    k_pack_b<<<(G3 * HH + 255) / 256, 256, 0, stream>>>(Whh0, whh0p, G3, HH);
    k_pack_b<<<(G3 * HH + 255) / 256, 256, 0, stream>>>(Whh1, whh1p, G3, HH);
    k_pack_b<<<(G3 * HH + 255) / 256, 256, 0, stream>>>(Wih1, wih1p, G3, HH);
    k_pack_b<<<(HH * HH + 255) / 256, 256, 0, stream>>>(W1,   w1p,   HH, HH);

    // layer 0 input gates (K=16, memory bound)
    k_xg0<<<(BT * G3) / 256, 256, 0, stream>>>(x, Wih0, bih0, xg);

    // layer 0 scan -> h1 (4 WGs, one per 16 batch rows)
    k_scan<<<4, 512, SCAN_LDS, stream>>>(xg, whh0p, bhh0, h1);

    // layer 1 input gates: xg = h1 @ Wih1^T + bih1 (WMMA bulk GEMM)
    {
        dim3 grid(BT / 256, G3 / 32);
        gemm_f32<<<grid, 256, 256 * 256 * 2, stream>>>(h1, wih1p, bih1, xg, BT, G3, HH);
    }

    // layer 1 scan -> h2
    k_scan<<<4, 512, SCAN_LDS, stream>>>(xg, whh1p, bhh1, h2);

    // regressor: r = relu(h2 @ W1^T + b1)
    {
        dim3 grid(BT / 256, HH / 32);
        gemm_relu<<<grid, 256, 256 * 256 * 2, stream>>>(h2, w1p, b1, rbuf, BT, HH, HH);
    }

    // head: out = relu(r @ W2^T + b2)
    k_head<<<BT / 8, 256, 0, stream>>>(rbuf, W2, b2, out, BT);
}
